// RRF_GRU_Learnable_Delays_12644383719535
// MI455X (gfx1250) — compile-verified, matrix operand order checked by reading, not run on hardware
//
#include <hip/hip_runtime.h>
#include <stdint.h>

#define F_IN   34
#define KW     7
#define STR    3
#define FD     10
#define C_CH   64
#define H_DIM  640
#define H3     1920
#define T_MAXK 9
#define PADW   4
#define B_SZ   32
#define T_LEN  1000
#define KB_N   20      /* 640 / 32 K-blocks  */
#define CN_N   120     /* 1920 / 16 col tiles */
#define BN_CNT (T_LEN * B_SZ * FD)   /* 320000 */
#define BN_EPSF 1e-5f

typedef __attribute__((ext_vector_type(16))) __bf16        v16bf;
typedef __attribute__((ext_vector_type(8)))  float         v8f;
typedef __attribute__((ext_vector_type(8)))  unsigned int  v8u;

static __device__ __forceinline__ unsigned short f2bf(float f) {
  unsigned int u = __builtin_bit_cast(unsigned int, f);
  unsigned int r = (u + 0x7FFFu + ((u >> 16) & 1u)) >> 16;   // RNE
  return (unsigned short)r;
}
static __device__ __forceinline__ float sigmoidf(float x) {
  return 1.0f / (1.0f + __expf(-x));
}
static __device__ __forceinline__ v16bf ld_bf16x16(const __bf16* p0,
                                                   const __bf16* p1) {
  uint4 a = *(const uint4*)p0;
  uint4 b = *(const uint4*)p1;
  union { v8u u; v16bf v; } r;
  r.u = (v8u){a.x, a.y, a.z, a.w, b.x, b.y, b.z, b.w};
  return r.v;
}

// ---------------------------------------------------------------------------
// 0) init: zero h state (f32 + bf16 mirror) and BN accumulators
// ---------------------------------------------------------------------------
__global__ void init_ws(float* __restrict__ h, __bf16* __restrict__ h_bf,
                        float* __restrict__ bnsum, float* __restrict__ bnsq) {
  int i = blockIdx.x * blockDim.x + threadIdx.x;
  if (i < B_SZ * H_DIM) {
    h[i] = 0.0f;
    h_bf[i] = __builtin_bit_cast(__bf16, (unsigned short)0);
  }
  if (i < C_CH) { bnsum[i] = 0.0f; bnsq[i] = 0.0f; }
}

// ---------------------------------------------------------------------------
// 1) DCLS kernel materialization: dk[ho][ci][tau] = w * relu(1-|tau-clip(p)|)
// ---------------------------------------------------------------------------
__global__ void build_dk(const float* __restrict__ dcls_w,
                         const float* __restrict__ dcls_p,
                         float* __restrict__ dk) {
  int i = blockIdx.x * blockDim.x + threadIdx.x;
  if (i >= H_DIM * 2 * KW * T_MAXK) return;
  int tau = i % T_MAXK;
  int rc  = i / T_MAXK;                 // rc = ho*14 + ci
  float p = dcls_p[rc];
  p = fminf(fmaxf(p, 0.0f), (float)(T_MAXK - 1));
  float tri = 1.0f - fabsf((float)tau - p);
  dk[i] = dcls_w[rc] * fmaxf(tri, 0.0f);
}

// ---------------------------------------------------------------------------
// 2) EMA low-pass (recurrent form of the 200-tap geometric kernel) + high-pass
// ---------------------------------------------------------------------------
__global__ void ema_highpass(const float* __restrict__ x,
                             const float* __restrict__ a_vals,
                             const float* __restrict__ w_vals,
                             float* __restrict__ hp) {
  int i = blockIdx.x * blockDim.x + threadIdx.x;
  if (i >= B_SZ * F_IN) return;
  int f = i % F_IN;
  const float a = a_vals[f];
  const float w = w_vals[f];
  const float* xr = x + (long)i * T_LEN;
  float*       hr = hp + (long)i * T_LEN;
  float low = 0.0f;
  for (int t = 0; t < T_LEN; ++t) {
    float xv = xr[t];
    low = a * xv + (1.0f - a) * low;
    hr[t] = xv - w * low;
  }
}

// ---------------------------------------------------------------------------
// 3) DCLS grouped conv + per-C BN partial sums
// ---------------------------------------------------------------------------
__global__ void __launch_bounds__(256)
dcls_conv_bnstats(const float* __restrict__ hp, const float* __restrict__ dk,
                  const float* __restrict__ dcls_b, float* __restrict__ y_raw,
                  float* __restrict__ bnsum, float* __restrict__ bnsq) {
  __shared__ float lsum[C_CH], lsq[C_CH];
  int tid = threadIdx.x;
  if (tid < C_CH) { lsum[tid] = 0.0f; lsq[tid] = 0.0f; }
  __syncthreads();

  long idx = (long)blockIdx.x * 256 + tid;
  if (idx < (long)T_LEN * B_SZ * H_DIM) {
    int  ho = (int)(idx % H_DIM);
    long tb = idx / H_DIM;
    int  b  = (int)(tb % B_SZ);
    int  t  = (int)(tb / B_SZ);
    int  fd = ho / C_CH;
    float acc = dcls_b[ho];
    const float* dkr = dk + (long)ho * (2 * KW) * T_MAXK;
    #pragma unroll
    for (int ci = 0; ci < 2 * KW; ++ci) {
      int s = ci / KW, k = ci % KW;
      const float* hpr = hp + ((long)b * F_IN + fd * STR + k) * T_LEN;
      #pragma unroll
      for (int tau = 0; tau < T_MAXK; ++tau) {
        int tt = t + tau - PADW;
        if (tt >= 0 && tt < T_LEN) {
          float v = hpr[tt];
          v = s ? fmaxf(-v, 0.0f) : fmaxf(v, 0.0f);
          acc += v * dkr[ci * T_MAXK + tau];
        }
      }
    }
    y_raw[idx] = acc;
    int c = ho % C_CH;
    atomicAdd(&lsum[c], acc);
    atomicAdd(&lsq[c], acc * acc);
  }
  __syncthreads();
  if (tid < C_CH) {
    atomicAdd(&bnsum[tid], lsum[tid]);
    atomicAdd(&bnsq[tid],  lsq[tid]);
  }
}

// ---------------------------------------------------------------------------
// 4) BN finalize + sigmoid + channel reshuffle (h = c*10 + fd) + bf16 pack
// ---------------------------------------------------------------------------
__global__ void bn_sigmoid_pack(const float* __restrict__ y_raw,
                                const float* __restrict__ bnsum,
                                const float* __restrict__ bnsq,
                                const float* __restrict__ gamma,
                                const float* __restrict__ beta,
                                __bf16* __restrict__ seq_bf) {
  long idx = (long)blockIdx.x * 256 + threadIdx.x;
  if (idx >= (long)T_LEN * B_SZ * H_DIM) return;
  int  ho = (int)(idx % H_DIM);
  long tb = idx / H_DIM;
  int  c  = ho % C_CH;
  int  fd = ho / C_CH;
  const float inv_n = 1.0f / (float)BN_CNT;
  float mean = bnsum[c] * inv_n;
  float var  = bnsq[c] * inv_n - mean * mean;
  float yv = (y_raw[idx] - mean) * rsqrtf(var + BN_EPSF);
  float sv = sigmoidf(yv * gamma[c] + beta[c]);
  int hseq = c * FD + fd;
  unsigned short s = f2bf(sv);
  seq_bf[tb * H_DIM + hseq] = __builtin_bit_cast(__bf16, s);
}

// ---------------------------------------------------------------------------
// 5) Pack W (1920x640 f32, row-major) into WMMA B-operand bf16 layout:
//    out[((cn*20 + kb)*32 + lane)*16 + e], e=2v+p maps to
//    k = kb*32 + (v/4)*16 + (lane/16)*8 + (v%4)*2 + p ; n = cn*16 + lane%16
// ---------------------------------------------------------------------------
__global__ void pack_weight(const float* __restrict__ W, __bf16* __restrict__ out) {
  int idx = blockIdx.x * blockDim.x + threadIdx.x;
  if (idx >= CN_N * KB_N * 32) return;
  int lane = idx & 31;
  int rest = idx >> 5;
  int kb = rest % KB_N;
  int cn = rest / KB_N;
  int nloc = lane & 15, half = lane >> 4;
  int n = cn * 16 + nloc;
  __bf16* o = out + (long)idx * 16;
  #pragma unroll
  for (int e = 0; e < 16; ++e) {
    int v = e >> 1, p = e & 1;
    int k = kb * 32 + (v >> 2) * 16 + half * 8 + (v & 3) * 2 + p;
    unsigned short s = f2bf(W[(long)n * H_DIM + k]);
    o[e] = __builtin_bit_cast(__bf16, s);
  }
}

// ---------------------------------------------------------------------------
// 6) GRU step GEMM, 2x2-tile fused, K loop fully unrolled (20 K-blocks):
//    straight-line code so every WMMA accumulates in place (VDST==SRC2) —
//    no loop-carried register rotation, no WMMA->VALU hazard NOP blocks.
// ---------------------------------------------------------------------------
__global__ void __launch_bounds__(32)
gru_step_gemm(const __bf16* __restrict__ seq_bf,
              const __bf16* __restrict__ h_bf,
              const __bf16* __restrict__ wih_pk,
              const __bf16* __restrict__ whh_pk,
              const float* __restrict__ b_ih,
              const float* __restrict__ b_hh,
              float* __restrict__ gx,
              float* __restrict__ gh,
              int t) {
  const int cn0  = blockIdx.x * 2;   // two column tiles: cn0, cn0+1
  const int lane = threadIdx.x;
  const int m    = lane & 15;
  const int half = lane >> 4;

  const long srow0 = ((long)t * B_SZ + m) * H_DIM;       // rows b = m
  const long srow1 = srow0 + 16L * H_DIM;                // rows b = 16 + m
  const long hrow0 = (long)m * H_DIM;
  const long hrow1 = hrow0 + 16L * H_DIM;

  v8f accx[2][2] = {{{}, {}}, {{}, {}}};   // [rb][cj]  seq @ W_ih^T
  v8f acch[2][2] = {{{}, {}}, {{}, {}}};   // [rb][cj]  h   @ W_hh^T

  #pragma unroll
  for (int kb = 0; kb < KB_N; ++kb) {
    const int k0 = kb * 32 + half * 8;
    const int k1 = k0 + 16;

    // A operands (bf16), both row tiles
    v16bf ax0 = ld_bf16x16(seq_bf + srow0 + k0, seq_bf + srow0 + k1);
    v16bf ax1 = ld_bf16x16(seq_bf + srow1 + k0, seq_bf + srow1 + k1);
    v16bf ah0 = ld_bf16x16(h_bf + hrow0 + k0, h_bf + hrow0 + k1);
    v16bf ah1 = ld_bf16x16(h_bf + hrow1 + k0, h_bf + hrow1 + k1);

    // B operands (pre-packed), both column tiles, both weight matrices
    const long b0 = (((long)cn0 * KB_N + kb) * 32 + lane) * 16;
    const long b1 = b0 + (long)KB_N * 32 * 16;           // cn0+1 strip
    v16bf bx0 = ld_bf16x16(wih_pk + b0, wih_pk + b0 + 8);
    v16bf bx1 = ld_bf16x16(wih_pk + b1, wih_pk + b1 + 8);
    v16bf bh0 = ld_bf16x16(whh_pk + b0, whh_pk + b0 + 8);
    v16bf bh1 = ld_bf16x16(whh_pk + b1, whh_pk + b1 + 8);

    accx[0][0] = __builtin_amdgcn_wmma_f32_16x16x32_bf16(
        false, ax0, false, bx0, (short)0, accx[0][0], false, false);
    accx[0][1] = __builtin_amdgcn_wmma_f32_16x16x32_bf16(
        false, ax0, false, bx1, (short)0, accx[0][1], false, false);
    accx[1][0] = __builtin_amdgcn_wmma_f32_16x16x32_bf16(
        false, ax1, false, bx0, (short)0, accx[1][0], false, false);
    accx[1][1] = __builtin_amdgcn_wmma_f32_16x16x32_bf16(
        false, ax1, false, bx1, (short)0, accx[1][1], false, false);
    acch[0][0] = __builtin_amdgcn_wmma_f32_16x16x32_bf16(
        false, ah0, false, bh0, (short)0, acch[0][0], false, false);
    acch[0][1] = __builtin_amdgcn_wmma_f32_16x16x32_bf16(
        false, ah0, false, bh1, (short)0, acch[0][1], false, false);
    acch[1][0] = __builtin_amdgcn_wmma_f32_16x16x32_bf16(
        false, ah1, false, bh0, (short)0, acch[1][0], false, false);
    acch[1][1] = __builtin_amdgcn_wmma_f32_16x16x32_bf16(
        false, ah1, false, bh1, (short)0, acch[1][1], false, false);
  }

  // D layout (32-bit C/D 16x16): lane -> N, VGPR r -> M = r + 8*half
  #pragma unroll
  for (int cj = 0; cj < 2; ++cj) {
    const int j = (cn0 + cj) * 16 + m;
    const float bi  = b_ih[j];
    const float bh2 = b_hh[j];
    #pragma unroll
    for (int rb = 0; rb < 2; ++rb) {
      #pragma unroll
      for (int r = 0; r < 8; ++r) {
        const int brow = rb * 16 + half * 8 + r;
        gx[(long)brow * H3 + j] = accx[rb][cj][r] + bi;
        gh[(long)brow * H3 + j] = acch[rb][cj][r] + bh2;
      }
    }
  }
}

// ---------------------------------------------------------------------------
// 7) GRU gate nonlinearity + state update (f32 + bf16 mirror) + fc reduction
// ---------------------------------------------------------------------------
__global__ void __launch_bounds__(H_DIM)
gru_gate(const float* __restrict__ gx, const float* __restrict__ gh,
         const float* __restrict__ fc_w, const float* __restrict__ fc_b,
         float* __restrict__ h, __bf16* __restrict__ h_bf,
         float* __restrict__ out, int t) {
  __shared__ float red[H_DIM];
  const int b = blockIdx.x;
  const int j = threadIdx.x;
  const float xr = gx[(long)b * H3 + j];
  const float hr = gh[(long)b * H3 + j];
  const float xz = gx[(long)b * H3 + H_DIM + j];
  const float hz = gh[(long)b * H3 + H_DIM + j];
  const float xn = gx[(long)b * H3 + 2 * H_DIM + j];
  const float hn = gh[(long)b * H3 + 2 * H_DIM + j];
  const float r = sigmoidf(xr + hr);
  const float z = sigmoidf(xz + hz);
  const float n = tanhf(xn + r * hn);
  const float hv = (1.0f - z) * n + z * h[(long)b * H_DIM + j];
  h[(long)b * H_DIM + j] = hv;
  h_bf[(long)b * H_DIM + j] = __builtin_bit_cast(__bf16, f2bf(hv));
  red[j] = hv * fc_w[j];
  __syncthreads();
  for (int s2 = 512; s2 > 0; s2 >>= 1) {
    if (j < s2 && j + s2 < H_DIM) red[j] += red[j + s2];
    __syncthreads();
  }
  if (j == 0) out[(long)b * T_LEN + t] = red[0] + fc_b[0];
}

// ---------------------------------------------------------------------------
// launch
// ---------------------------------------------------------------------------
extern "C" void kernel_launch(void* const* d_in, const int* in_sizes, int n_in,
                              void* d_out, int out_size, void* d_ws, size_t ws_size,
                              hipStream_t stream) {
  const float* x        = (const float*)d_in[0];
  const float* a_vals   = (const float*)d_in[1];
  const float* w_vals   = (const float*)d_in[2];
  const float* dcls_w   = (const float*)d_in[3];
  const float* dcls_p   = (const float*)d_in[4];
  const float* dcls_b   = (const float*)d_in[5];
  const float* bn_gamma = (const float*)d_in[6];
  const float* bn_beta  = (const float*)d_in[7];
  const float* W_ih     = (const float*)d_in[8];
  const float* W_hh     = (const float*)d_in[9];
  const float* b_ih     = (const float*)d_in[10];
  const float* b_hh     = (const float*)d_in[11];
  const float* fc_w     = (const float*)d_in[12];
  const float* fc_b     = (const float*)d_in[13];
  float* out = (float*)d_out;

  // workspace carve (256B aligned)
  char* wp = (char*)d_ws;
  auto alloc = [&](size_t bytes) -> char* {
    char* p = wp;
    wp += (bytes + 255) & ~(size_t)255;
    return p;
  };
  const long NTOT = (long)T_LEN * B_SZ * H_DIM;     // 20.48M
  float*  hp      = (float*)alloc(sizeof(float) * B_SZ * F_IN * T_LEN);
  float*  dk      = (float*)alloc(sizeof(float) * H_DIM * 2 * KW * T_MAXK);
  float*  y_raw   = (float*)alloc(sizeof(float) * (size_t)NTOT);
  __bf16* seq_bf  = (__bf16*)alloc(sizeof(unsigned short) * (size_t)NTOT);
  __bf16* wih_pk  = (__bf16*)alloc(sizeof(unsigned short) * (size_t)H3 * H_DIM);
  __bf16* whh_pk  = (__bf16*)alloc(sizeof(unsigned short) * (size_t)H3 * H_DIM);
  float*  hbuf    = (float*)alloc(sizeof(float) * B_SZ * H_DIM);
  __bf16* hbf     = (__bf16*)alloc(sizeof(unsigned short) * B_SZ * H_DIM);
  float*  gxb     = (float*)alloc(sizeof(float) * B_SZ * H3);
  float*  ghb     = (float*)alloc(sizeof(float) * B_SZ * H3);
  float*  bnsum   = (float*)alloc(sizeof(float) * C_CH);
  float*  bnsq    = (float*)alloc(sizeof(float) * C_CH);

  init_ws<<<(B_SZ * H_DIM + 255) / 256, 256, 0, stream>>>(hbuf, hbf, bnsum, bnsq);
  build_dk<<<(H_DIM * 2 * KW * T_MAXK + 255) / 256, 256, 0, stream>>>(dcls_w, dcls_p, dk);
  ema_highpass<<<(B_SZ * F_IN + 255) / 256, 256, 0, stream>>>(x, a_vals, w_vals, hp);
  dcls_conv_bnstats<<<(int)((NTOT + 255) / 256), 256, 0, stream>>>(hp, dk, dcls_b, y_raw, bnsum, bnsq);
  bn_sigmoid_pack<<<(int)((NTOT + 255) / 256), 256, 0, stream>>>(y_raw, bnsum, bnsq, bn_gamma, bn_beta, seq_bf);
  pack_weight<<<(CN_N * KB_N * 32 + 255) / 256, 256, 0, stream>>>(W_ih, wih_pk);
  pack_weight<<<(CN_N * KB_N * 32 + 255) / 256, 256, 0, stream>>>(W_hh, whh_pk);

  for (int t = 0; t < T_LEN; ++t) {
    gru_step_gemm<<<CN_N / 2, 32, 0, stream>>>(seq_bf, hbf, wih_pk, whh_pk,
                                               b_ih, b_hh, gxb, ghb, t);
    gru_gate<<<B_SZ, H_DIM, 0, stream>>>(gxb, ghb, fc_w, fc_b, hbuf, hbf, out, t);
  }
}
